// EdgeDecoder_11003706212841
// MI455X (gfx1250) — matrix operand order repaired and checked by reading, not compiled, not run on hardware
//
#include <hip/hip_runtime.h>
#include <math.h>

// Shapes fixed by setup_inputs()
#define BDIM 4
#define NDIM 512
#define DDIM 256
#define HDIM 128
#define ITER 8              // i-groups per block in the fused kernel

typedef __attribute__((ext_vector_type(16))) _Float16 v16h;
typedef __attribute__((ext_vector_type(8)))  _Float16 v8h;
typedef __attribute__((ext_vector_type(8)))  float    v8f;

// GELU(x) = 0.5 x (1 + erf(x/sqrt(2)))
// erf via Abramowitz-Stegun 7.1.26 (|abs err| < 1.5e-7), raw v_rcp/v_exp.
__device__ __forceinline__ float gelu_fast(float x) {
    const float u  = fabsf(x) * 0.7071067811865476f;      // |x|/sqrt2
    const float t  = __builtin_amdgcn_rcpf(fmaf(0.3275911f, u, 1.0f));
    float p = fmaf(1.061405429f, t, -1.453152027f);
    p = fmaf(p, t, 1.421413741f);
    p = fmaf(p, t, -0.284496736f);
    p = fmaf(p, t, 0.254829592f);
    p = p * t;
    // exp(-u^2) = exp2(-u^2 * log2(e))
    const float e    = __builtin_amdgcn_exp2f(u * u * -1.4426950408889634f);
    const float erfu = fmaf(-p, e, 1.0f);                 // erf(u), u >= 0
    const float s    = copysignf(erfu, x);                // odd symmetry
    return 0.5f * x * (1.0f + s);
}

// ---------------------------------------------------------------------------
// Kernel 1: fold projection weights through W1 halves.
//   Wc_src = W_src @ W1[:H]   (D x H)
//   Wc_tgt = W_tgt @ W1[H:]   (D x H)
//   bfuse  = b1 + b_src @ W1[:H] + b_tgt @ W1[H:]
// grid (D+1, 2), block H
// ---------------------------------------------------------------------------
__global__ void fold_weights_kernel(const float* __restrict__ W_src,
                                    const float* __restrict__ b_src,
                                    const float* __restrict__ W_tgt,
                                    const float* __restrict__ b_tgt,
                                    const float* __restrict__ W1,
                                    const float* __restrict__ b1,
                                    float* __restrict__ Wc_src,
                                    float* __restrict__ Wc_tgt,
                                    float* __restrict__ bfuse) {
    const int r = blockIdx.x;     // row of D, or D => bias job
    const int y = blockIdx.y;     // 0 = src, 1 = tgt
    const int c = threadIdx.x;    // col of H

    if (r < DDIM) {
        const float* W  = y ? W_tgt : W_src;
        float*       Wc = y ? Wc_tgt : Wc_src;
        float acc = 0.0f;
        for (int k = 0; k < HDIM; ++k)
            acc += W[r * HDIM + k] * W1[(k + y * HDIM) * HDIM + c];
        Wc[r * HDIM + c] = acc;
    } else if (y == 0) {
        float acc = b1[c];
        for (int k = 0; k < HDIM; ++k) {
            acc += b_src[k] * W1[k * HDIM + c];
            acc += b_tgt[k] * W1[(k + HDIM) * HDIM + c];
        }
        bfuse[c] = acc;
    }
}

// ---------------------------------------------------------------------------
// Kernel 2: src1 = E @ Wc_src ; tgt1 = E @ Wc_tgt  (bias folded into bfuse)
// grid (B*N, 2), block H
// ---------------------------------------------------------------------------
__global__ void proj_kernel(const float* __restrict__ E,
                            const float* __restrict__ Wc_src,
                            const float* __restrict__ Wc_tgt,
                            float* __restrict__ src1,
                            float* __restrict__ tgt1) {
    __shared__ float eRow[DDIM];
    const int row = blockIdx.x;
    const int y   = blockIdx.y;
    const int c   = threadIdx.x;

    for (int idx = threadIdx.x; idx < DDIM; idx += HDIM)
        eRow[idx] = E[(size_t)row * DDIM + idx];
    __syncthreads();

    const float* W = y ? Wc_tgt : Wc_src;
    float*       o = y ? tgt1   : src1;
    float acc = 0.0f;
    #pragma unroll 8
    for (int k = 0; k < DDIM; ++k)
        acc += eRow[k] * W[k * HDIM + c];
    o[(size_t)row * HDIM + c] = acc;
}

// ---------------------------------------------------------------------------
// Kernel 3: fused pairwise MLP with f16 WMMA.
//   out[b,i,j] = (i==j) ? 0 :
//     gelu(gelu(src1_i + tgt1_j + bfuse) @ W2 + b2) @ W3 + b3
//
// grid (N/16, N/(8*ITER), B), block 256 (8 waves).
// W2 (f16, transposed) + tgt rows staged once per block, then the block
// iterates over ITER i-groups; wave w handles i = i0 + 8*it + w, all waves
// share the 16-wide j tile.
// ---------------------------------------------------------------------------
#define W2T_STRIDE 136   // f16 elements; 68 dwords -> bank-conflict-free

__global__ void __launch_bounds__(256)
edge_fused_kernel(const float* __restrict__ src1,
                  const float* __restrict__ tgt1,
                  const float* __restrict__ bfuse,
                  const float* __restrict__ W2,
                  const float* __restrict__ b2,
                  const float* __restrict__ W3,
                  const float* __restrict__ pb3,
                  float* __restrict__ out) {
    __shared__ __align__(16) _Float16 w2t[HDIM * W2T_STRIDE]; // W2^T: [col][k]
    __shared__ float trow[16 * 132];                          // tgt1 rows, padded
    __shared__ float srow[8 * HDIM];                          // src1_i + bfuse / wave
    __shared__ float sb2[HDIM];
    __shared__ float sw3[HDIM];
    __shared__ float sbf[HDIM];

    const int tid  = threadIdx.x;
    const int lane = tid & 31;
    const int wave = tid >> 5;
    const int half = lane >> 4;   // 0/1
    const int m    = lane & 15;

    const int j0 = blockIdx.x * 16;
    const int i0 = blockIdx.y * (8 * ITER);
    const int b  = blockIdx.z;

    // --- one-time staging: W2^T (f16), tgt rows, vectors -------------------
    for (int idx = tid; idx < HDIM * HDIM; idx += 256) {
        int k = idx >> 7, c = idx & (HDIM - 1);
        w2t[c * W2T_STRIDE + k] = (_Float16)W2[k * HDIM + c];
    }
    for (int idx = tid; idx < 16 * HDIM; idx += 256) {
        int r = idx >> 7, k = idx & (HDIM - 1);
        trow[r * 132 + k] = tgt1[((size_t)b * NDIM + (j0 + r)) * HDIM + k];
    }
    if (tid < HDIM) { sb2[tid] = b2[tid]; sw3[tid] = W3[tid]; sbf[tid] = bfuse[tid]; }
    const float b3v = *pb3;
    __syncthreads();

    const float* sr = &srow[wave * HDIM];
    const float* tr = &trow[m * 132];

    for (int it = 0; it < ITER; ++it) {
        const int i = i0 + 8 * it + wave;

        // stage this i-group's src rows (+fused bias)
        __syncthreads();   // previous iteration finished reading srow
        for (int idx = tid; idx < 8 * HDIM; idx += 256) {
            int w = idx >> 7, k = idx & (HDIM - 1);
            srow[w * HDIM + k] =
                src1[((size_t)b * NDIM + (i0 + 8 * it + w)) * HDIM + k] + sbf[k];
        }
        __syncthreads();

        // --- A fragments: X[j][k] = gelu(src_i[k] + tgt_j[k] + bias[k]) ----
        // A 16x32 f16 lane layout (ISA 7.12.2): lane row = lane&15,
        // element e -> K = 32c + 16*(v>>2) + 2*(v&3) + (e&1) + 8*half, v=e>>1
        v16h afrag[4];
        #pragma unroll
        for (int c = 0; c < 4; ++c) {
            v16h a;
            #pragma unroll
            for (int e = 0; e < 16; ++e) {
                const int v   = e >> 1;
                const int k   = 32 * c + 16 * (v >> 2) + 2 * (v & 3) + (e & 1) + 8 * half;
                const float x = sr[k] + tr[k];
                a[e] = (_Float16)gelu_fast(x);
            }
            afrag[c] = a;
        }

        // --- GEMM vs W2, fused second GELU + W3 dot ------------------------
        float partial[8];
        #pragma unroll
        for (int r = 0; r < 8; ++r) partial[r] = 0.0f;

        #pragma unroll
        for (int nt = 0; nt < 8; ++nt) {
            const int col = nt * 16 + m;        // this lane's h-column
            v8f acc = {};
            #pragma unroll
            for (int c = 0; c < 4; ++c) {
                // B 32x16 f16: lane col = lane&15, K = 32c + 8*half + (0..7), 32c + 16 + 8*half + (0..7)
                const _Float16* wp = &w2t[col * W2T_STRIDE + c * 32 + 8 * half];
                v8h lo = *(const v8h*)(wp);
                v8h hi = *(const v8h*)(wp + 16);
                v16h bf;
                #pragma unroll
                for (int e = 0; e < 8; ++e) { bf[e] = lo[e]; bf[e + 8] = hi[e]; }
                acc = __builtin_amdgcn_wmma_f32_16x16x32_f16(
                    /*neg_a=*/false, afrag[c], /*neg_b=*/false, bf,
                    /*c_mod=*/(short)0, acc, /*reuse_a=*/false, /*reuse_b=*/false);
            }
            // acc[r] = Y[j = r + 8*half][col]
            const float b2v = sb2[col];
            const float w3v = sw3[col];
            #pragma unroll
            for (int r = 0; r < 8; ++r)
                partial[r] += w3v * gelu_fast(acc[r] + b2v);
        }

        // --- reduce across the 16 lanes of each half-group -----------------
        #pragma unroll
        for (int r = 0; r < 8; ++r) {
            float v = partial[r];
            v += __shfl_xor(v, 1);
            v += __shfl_xor(v, 2);
            v += __shfl_xor(v, 4);
            v += __shfl_xor(v, 8);
            partial[r] = v;   // now group-uniform
        }

        if (m < 8) {
            // select partial[m] without dynamic indexing (avoid scratch)
            float v = partial[0];
            #pragma unroll
            for (int r = 1; r < 8; ++r) v = (m == r) ? partial[r] : v;
            const int j = j0 + m + 8 * half;
            const float res = (i == j) ? 0.0f : (v + b3v);
            out[((size_t)b * NDIM + i) * NDIM + j] = res;
        }
    }
}

// ---------------------------------------------------------------------------
extern "C" void kernel_launch(void* const* d_in, const int* in_sizes, int n_in,
                              void* d_out, int out_size, void* d_ws, size_t ws_size,
                              hipStream_t stream) {
    const float* E     = (const float*)d_in[0];
    const float* W_src = (const float*)d_in[1];
    const float* b_src = (const float*)d_in[2];
    const float* W_tgt = (const float*)d_in[3];
    const float* b_tgt = (const float*)d_in[4];
    const float* W1    = (const float*)d_in[5];
    const float* b1    = (const float*)d_in[6];
    const float* W2    = (const float*)d_in[7];
    const float* b2    = (const float*)d_in[8];
    const float* W3    = (const float*)d_in[9];
    const float* b3    = (const float*)d_in[10];
    float* out = (float*)d_out;

    // workspace carve-up (floats)
    float* ws      = (float*)d_ws;
    float* Wc_src  = ws;                              // D*H
    float* Wc_tgt  = Wc_src + DDIM * HDIM;            // D*H
    float* bfuse   = Wc_tgt + DDIM * HDIM;            // H
    float* src1    = bfuse + HDIM;                    // B*N*H
    float* tgt1    = src1 + (size_t)BDIM * NDIM * HDIM;

    fold_weights_kernel<<<dim3(DDIM + 1, 2), HDIM, 0, stream>>>(
        W_src, b_src, W_tgt, b_tgt, W1, b1, Wc_src, Wc_tgt, bfuse);

    proj_kernel<<<dim3(BDIM * NDIM, 2), HDIM, 0, stream>>>(
        E, Wc_src, Wc_tgt, src1, tgt1);

    edge_fused_kernel<<<dim3(NDIM / 16, NDIM / (8 * ITER), BDIM), 256, 0, stream>>>(
        src1, tgt1, bfuse, W2, b2, W3, b3, out);
}
